// FPLPGCN_dw_linear_1168231104607
// MI455X (gfx1250) — matrix operand order repaired and checked by reading, not compiled, not run on hardware
//
#include <hip/hip_runtime.h>
#include <hip/hip_bf16.h>

// ---------------------------------------------------------------------------
// FPLP-GCN on MI455X (gfx1250): bf16 WMMA GEMMs + L2-resident atomic scatter
// ---------------------------------------------------------------------------
// Sizes (fixed by reference): N=50000, E=800000, IN=128, HID=96, OUT=40,
// DW=64, N_LBL=10.  M=50000 = 3125 * 16 -> exact 16-row tiles.

typedef __attribute__((ext_vector_type(16))) __bf16 v16bf;
typedef __attribute__((ext_vector_type(8)))  float  v8f;

// native fptrunc (RNE) -> lets the backend use packed bf16 cvt instructions
__device__ __forceinline__ __bf16 tobf(float f) { return (__bf16)f; }

// ---------------------------- small helpers --------------------------------

__global__ void k_zero(float* __restrict__ p, int n) {
    int i = blockIdx.x * blockDim.x + threadIdx.x;
    if (i < n) p[i] = 0.f;
}

__global__ void k_count_deg(const int* __restrict__ dst, float* __restrict__ deg, int E) {
    int e = blockIdx.x * blockDim.x + threadIdx.x;
    if (e < E) atomicAdd(&deg[dst[e]], 1.0f);
}

__global__ void k_dinv(float* __restrict__ dinv, int n) {
    int i = blockIdx.x * blockDim.x + threadIdx.x;
    if (i < n) dinv[i] = rsqrtf(dinv[i] + 1.0f);   // deg+1 > 0 always
}

__global__ void k_norm(const int* __restrict__ src, const int* __restrict__ dst,
                       const float* __restrict__ dinv, float* __restrict__ norm, int E) {
    int e = blockIdx.x * blockDim.x + threadIdx.x;
    if (e < E) norm[e] = dinv[src[e]] * dinv[dst[e]];
}

// Pack weight W[K,NCOL] (row-major fp32) into per-lane bf16 B fragments:
// Bp[((kt*NT+nt)*32 + lane)*16 + i]  <->  K = kt*32 + (lane>=16?8:0) + (i&7) + (i>=8?16:0),
//                                         n = nt*16 + (lane&15);  OOB -> 0.
__global__ void k_pack_w(const float* __restrict__ W, unsigned short* __restrict__ Bp,
                         int K, int NCOL, int KT, int NT) {
    int idx = blockIdx.x * blockDim.x + threadIdx.x;
    int total = KT * NT * 32 * 16;
    if (idx >= total) return;
    int i    = idx & 15;
    int lane = (idx >> 4) & 31;
    int t    = idx >> 9;
    int nt   = t % NT;
    int kt   = t / NT;
    int k = kt * 32 + ((lane >= 16) ? 8 : 0) + (i & 7) + ((i >= 8) ? 16 : 0);
    int n = nt * 16 + (lane & 15);
    float v = (k < K && n < NCOL) ? W[k * NCOL + n] : 0.f;
    Bp[idx] = __builtin_bit_cast(unsigned short, tobf(v));
}

// ------------------------------ WMMA GEMM ----------------------------------
// Out[M,NCOL] = A[M,K] @ W  (W pre-packed bf16).  One wave per 16x16 tile.
template<int K, int KT, int NCOL, int NT>
__global__ void k_gemm_wmma(const float* __restrict__ A,
                            const unsigned short* __restrict__ Bp,
                            float* __restrict__ Out, int M) {
    int wave = (int)((blockIdx.x * (long)blockDim.x + threadIdx.x) >> 5);
    int lane = threadIdx.x & 31;
    int mtiles = M >> 4;
    int nt = wave % NT;
    int mt = wave / NT;
    if (mt >= mtiles) return;
    int hi   = lane >> 4;
    int lsub = lane & 15;
    int row  = (mt << 4) + lsub;              // A-frag: M = lane%16
    const float* arow = A + (long)row * K;

    v8f acc = {};
    for (int kt = 0; kt < KT; ++kt) {
        int k0 = kt * 32 + hi * 8;
        v16bf a;
#pragma unroll
        for (int i = 0; i < 8; ++i) {
            int k1 = k0 + i;                  // elems 0..7  : K = k0..k0+7
            int k2 = k0 + 16 + i;             // elems 8..15 : K = k0+16..k0+23
            a[i]     = tobf((k1 < K) ? arow[k1] : 0.f);
            a[i + 8] = tobf((k2 < K) ? arow[k2] : 0.f);
        }
        v16bf b = *(const v16bf*)(Bp + (((long)kt * NT + nt) * 32 + lane) * 16);
        acc = __builtin_amdgcn_wmma_f32_16x16x32_bf16(false, a, false, b,
                                                      (short)0, acc, false, false);
    }
#pragma unroll
    for (int r = 0; r < 8; ++r) {             // C/D: vgpr r -> M=r+8*hi, N=lsub
        int n = nt * 16 + lsub;
        if (n < NCOL) {
            int rr = (mt << 4) + r + 8 * hi;
            Out[(long)rr * NCOL + n] = acc[r];
        }
    }
}

// Fused GEMM over concat([h96 | lab40 | dw64]) (K=200 pad 224) + bias + sigmoid.
__device__ __forceinline__ float fuse_fetch(const float* h96, const float* lab,
                                            const float* dw, int row, int k) {
    if (k < 96)  return h96[(long)row * 96 + k];
    if (k < 136) return lab[(long)row * 40 + (k - 96)];
    if (k < 200) return dw [(long)row * 64 + (k - 136)];
    return 0.f;
}

__global__ void k_fuse_wmma(const float* __restrict__ h96, const float* __restrict__ lab,
                            const float* __restrict__ dw,
                            const unsigned short* __restrict__ Bp,
                            const float* __restrict__ bias,
                            float* __restrict__ Out, int M) {
    const int KT = 7, NT = 3, NCOL = 40;
    int wave = (int)((blockIdx.x * (long)blockDim.x + threadIdx.x) >> 5);
    int lane = threadIdx.x & 31;
    int mtiles = M >> 4;
    int nt = wave % NT;
    int mt = wave / NT;
    if (mt >= mtiles) return;
    int hi   = lane >> 4;
    int lsub = lane & 15;
    int row  = (mt << 4) + lsub;

    v8f acc = {};
    for (int kt = 0; kt < KT; ++kt) {
        int k0 = kt * 32 + hi * 8;
        v16bf a;
#pragma unroll
        for (int i = 0; i < 8; ++i) {
            a[i]     = tobf(fuse_fetch(h96, lab, dw, row, k0 + i));
            a[i + 8] = tobf(fuse_fetch(h96, lab, dw, row, k0 + 16 + i));
        }
        v16bf b = *(const v16bf*)(Bp + (((long)kt * NT + nt) * 32 + lane) * 16);
        acc = __builtin_amdgcn_wmma_f32_16x16x32_bf16(false, a, false, b,
                                                      (short)0, acc, false, false);
    }
#pragma unroll
    for (int r = 0; r < 8; ++r) {
        int n = nt * 16 + lsub;
        if (n < NCOL) {
            int rr = (mt << 4) + r + 8 * hi;
            float v = acc[r] + bias[n];
            Out[(long)rr * NCOL + n] = 1.f / (1.f + __expf(-v));
        }
    }
}

// --------------------------- graph aggregation -----------------------------
// agg[n,f] = h[n,f]*dinv[n]^2 + b[f]   (self-loop + bias folded into init)
template<int F>
__global__ void k_init_agg(const float* __restrict__ h, const float* __restrict__ dinv,
                           const float* __restrict__ bias, float* __restrict__ agg, int n) {
    int idx = blockIdx.x * blockDim.x + threadIdx.x;
    if (idx >= n * F) return;
    int node = idx / F;
    int f    = idx - node * F;
    float di = dinv[node];
    agg[idx] = h[idx] * di * di + bias[f];
}

// agg[dst,f] += h[src,f] * norm[e]; float4 gather, 4 L2 atomics per thread.
template<int F>
__global__ void k_edge_agg(const int* __restrict__ src, const int* __restrict__ dst,
                           const float* __restrict__ norm, const float* __restrict__ h,
                           float* __restrict__ agg, int E) {
    const int Q = F / 4;
    int idx = blockIdx.x * blockDim.x + threadIdx.x;
    if (idx >= E * Q) return;
    int e = idx / Q;
    int q = (idx - e * Q) * 4;
    int s = src[e], d = dst[e];
    float w = norm[e];
    const float4 hv = *(const float4*)(h + (long)s * F + q);
    float* ap = agg + (long)d * F + q;
    atomicAdd(ap + 0, hv.x * w);
    atomicAdd(ap + 1, hv.y * w);
    atomicAdd(ap + 2, hv.z * w);
    atomicAdd(ap + 3, hv.w * w);
}

// ------------------------------- launcher ----------------------------------

static inline int cdiv(long a, long b) { return (int)((a + b - 1) / b); }

extern "C" void kernel_launch(void* const* d_in, const int* in_sizes, int n_in,
                              void* d_out, int out_size, void* d_ws, size_t ws_size,
                              hipStream_t stream) {
    const float* x   = (const float*)d_in[0];
    const float* y   = (const float*)d_in[1];
    const int*   ei  = (const int*)  d_in[2];
    const float* dwk = (const float*)d_in[3];
    const float* W1  = (const float*)d_in[4];
    const float* b1  = (const float*)d_in[5];
    const float* W2  = (const float*)d_in[6];
    const float* b2  = (const float*)d_in[7];
    const float* Wl  = (const float*)d_in[8];
    const float* bl  = (const float*)d_in[9];
    const float* Wf  = (const float*)d_in[10];
    const float* bfu = (const float*)d_in[11];

    const int N = in_sizes[0] / 128;
    const int E = in_sizes[2] / 2;
    const int* src = ei;
    const int* dst = ei + E;

    // ---- carve workspace (256B aligned) ----
    char* base = (char*)d_ws;
    size_t off = 0;
    auto carve = [&](size_t bytes) -> void* {
        void* p = base + off;
        off += (bytes + 255) & ~(size_t)255;
        return p;
    };
    float* dinv = (float*)carve((size_t)N * 4);          // also deg scratch
    float* nrm  = (float*)carve((size_t)E * 4);
    float* H    = (float*)carve((size_t)N * 96 * 4);     // gemm out (feat & label)
    float* A1   = (float*)carve((size_t)N * 96 * 4);
    float* A2   = (float*)carve((size_t)N * 96 * 4);
    float* LA   = (float*)carve((size_t)N * 40 * 4);
    float* LB   = (float*)carve((size_t)N * 40 * 4);
    unsigned short* BP1 = (unsigned short*)carve(4 * 6 * 512 * 2);   // 128->96
    unsigned short* BP2 = (unsigned short*)carve(3 * 6 * 512 * 2);   // 96->96
    unsigned short* BPL = (unsigned short*)carve(10 * 2 * 3 * 512 * 2); // 10x 40->40
    unsigned short* BPF = (unsigned short*)carve(7 * 3 * 512 * 2);   // 200->40

    const int T = 256;

    // ---- degree -> dinv -> edge norm (reused by all 12 layers) ----
    k_zero<<<cdiv(N, T), T, 0, stream>>>(dinv, N);
    k_count_deg<<<cdiv(E, T), T, 0, stream>>>(dst, dinv, E);
    k_dinv<<<cdiv(N, T), T, 0, stream>>>(dinv, N);
    k_norm<<<cdiv(E, T), T, 0, stream>>>(src, dst, dinv, nrm, E);

    // ---- pack weights to bf16 B fragments ----
    k_pack_w<<<cdiv(4 * 6 * 512, T), T, 0, stream>>>(W1, BP1, 128, 96, 4, 6);
    k_pack_w<<<cdiv(3 * 6 * 512, T), T, 0, stream>>>(W2, BP2, 96, 96, 3, 6);
    for (int i = 0; i < 10; ++i)
        k_pack_w<<<cdiv(2 * 3 * 512, T), T, 0, stream>>>(Wl + (size_t)i * 40 * 40,
                                                         BPL + (size_t)i * 2 * 3 * 512,
                                                         40, 40, 2, 3);
    k_pack_w<<<cdiv(7 * 3 * 512, T), T, 0, stream>>>(Wf, BPF, 200, 40, 7, 3);

    const int mtiles = N >> 4;                    // 3125 (exact)
    const int gw96 = cdiv((long)mtiles * 6 * 32, T);
    const int gw40 = cdiv((long)mtiles * 3 * 32, T);
    const int gi96 = cdiv((long)N * 96, T);
    const int gi40 = cdiv((long)N * 40, T);
    const int ge96 = cdiv((long)E * 24, T);       // F=96 -> 24 float4 chunks
    const int ge40 = cdiv((long)E * 10, T);       // F=40 -> 10 float4 chunks

    // ---- feature GCN layer 1: x[128] -> 96 ----
    k_gemm_wmma<128, 4, 96, 6><<<gw96, T, 0, stream>>>(x, BP1, H, N);
    k_init_agg<96><<<gi96, T, 0, stream>>>(H, dinv, b1, A1, N);
    k_edge_agg<96><<<ge96, T, 0, stream>>>(src, dst, nrm, H, A1, E);

    // ---- feature GCN layer 2: 96 -> 96 ----
    k_gemm_wmma<96, 3, 96, 6><<<gw96, T, 0, stream>>>(A1, BP2, H, N);
    k_init_agg<96><<<gi96, T, 0, stream>>>(H, dinv, b2, A2, N);
    k_edge_agg<96><<<ge96, T, 0, stream>>>(src, dst, nrm, H, A2, E);

    // ---- 10 label-propagation GCN layers: 40 -> 40 ----
    const float* cur = y;
    float* nxt = LA;
    for (int i = 0; i < 10; ++i) {
        k_gemm_wmma<40, 2, 40, 3><<<gw40, T, 0, stream>>>(
            cur, BPL + (size_t)i * 2 * 3 * 512, H, N);
        k_init_agg<40><<<gi40, T, 0, stream>>>(H, dinv, bl + (size_t)i * 40, nxt, N);
        k_edge_agg<40><<<ge40, T, 0, stream>>>(src, dst, nrm, H, nxt, E);
        cur = nxt;
        nxt = (nxt == LA) ? LB : LA;
    }

    // ---- fuse: sigmoid(concat(A2, labels, deepwalk) @ Wf + bf) ----
    k_fuse_wmma<<<gw40, T, 0, stream>>>(A2, cur, dwk, BPF, bfu, (float*)d_out, N);
}
